// SwinTransformer_81475529605565
// MI455X (gfx1250) — matrix operand order repaired and checked by reading, not compiled, not run on hardware
//
#include <hip/hip_runtime.h>
#include <hip/hip_bf16.h>

// Swin shifted-window attention, fused single kernel for MI455X (gfx1250).
// One workgroup (256 threads = 8 wave32) per window; all matmuls via
// v_wmma_f32_16x16x32_f16; intermediates in LDS (~145 KB of the 320 KB WGP pool).
// Weights are pre-converted to f16 into d_ws once per launch so the GEMM inner
// loops load B-fragments as raw b128s with zero f32->f16 VALU work.

#define WSZ 7
#define SSH 3
#define NHEAD 4
#define CCH 128
#define HH 56
#define WW 56
#define NTOK 49

#define QKVW_ELTS (384 * 128)
#define PROJW_ELTS (128 * 128)
#define WS_NEEDED ((QKVW_ELTS + PROJW_ELTS) * sizeof(_Float16))

typedef _Float16 v16h __attribute__((ext_vector_type(16)));
typedef _Float16 v8h  __attribute__((ext_vector_type(8)));
typedef float    v8f  __attribute__((ext_vector_type(8)));
typedef float    v4f  __attribute__((ext_vector_type(4)));

// ---- LDS layout (bytes) ----
#define OFF_X    0        // 64x128 f16  : shifted x window; reused later for attn-out O
#define OFF_QKV  16384    // 64x384 f16  : qkv projection result
#define OFF_VT   65536    // 4x32x64 f16 : V transposed per head (B operand of P*V)
#define OFF_S    81920    // 4x64x64 f32 : scores; softmax rewrites rows in-place as f16 P
#define OFF_RID  147456   // 64 int      : shift-mask region id per token
#define OFF_RS   147712   // 4x64 f32    : softmax row sums (normalization folded into O)
#define SMEM_BYTES 148736

__device__ __forceinline__ v8f wmma_f16(v16h a, v16h b, v8f c) {
  // D = A*B + C, 16x16x32, f16 in / f32 acc
  return __builtin_amdgcn_wmma_f32_16x16x32_f16(false, a, false, b, (short)0, c,
                                                false, false);
}

// A fragment: row = tile_row*16 + (lane&15); 16 contiguous halfs at chunk_base + (lane>>4)*16
// (caller passes k0 = chunk_base + (lane>>4)*8; matches B's contiguous k order)
__device__ __forceinline__ v16h ldsA(const _Float16* rowptr, int k0) {
  int kb = (k0 & ~15) + ((k0 >> 3) & 1) * 16;
  v8h lo = *(const v8h*)(rowptr + kb);
  v8h hi = *(const v8h*)(rowptr + kb + 8);
  v16h a;
#pragma unroll
  for (int i = 0; i < 8; ++i) { a[i] = lo[i]; a[i + 8] = hi[i]; }
  return a;
}

// B fragment: col = tile_col*16 + (lane&15); 16 contiguous k-halfs of B^T row
// starting at (lane>>4)*16 within the 32-k chunk. Works for LDS and global f16.
__device__ __forceinline__ v16h ldB16(const _Float16* p) {
  v8h lo = *(const v8h*)(p);
  v8h hi = *(const v8h*)(p + 8);
  v16h b;
#pragma unroll
  for (int i = 0; i < 8; ++i) { b[i] = lo[i]; b[i + 8] = hi[i]; }
  return b;
}

// Fallback: B fragment from global f32 weights (row-major W[o][c] == B^T), cvt to f16.
__device__ __forceinline__ v16h ldBf32(const float* p) {
  v16h b;
#pragma unroll
  for (int i = 0; i < 4; ++i) {
    v4f f = *(const v4f*)(p + i * 4);
#pragma unroll
    for (int j = 0; j < 4; ++j) b[i * 4 + j] = (_Float16)f[j];
  }
  return b;
}

// One-shot weight conversion: f32 -> f16 into workspace (qkv then proj, same
// row-major-W == B^T layout, so fragment loads stay 16-half contiguous).
__global__ __launch_bounds__(256)
void convert_weights_kernel(const float* __restrict__ qkvw,
                            const float* __restrict__ projw,
                            _Float16* __restrict__ wh) {
  int i = blockIdx.x * 256 + threadIdx.x;
  if (i < QKVW_ELTS) wh[i] = (_Float16)qkvw[i];
  if (i < PROJW_ELTS) wh[QKVW_ELTS + i] = (_Float16)projw[i];
}

template <bool F16W>
__global__ __launch_bounds__(256)
void swin_attn_kernel(const float* __restrict__ x,
                      const float* __restrict__ qkvw,   // [384,128]
                      const float* __restrict__ qkvb,   // [384]
                      const float* __restrict__ projw,  // [128,128]
                      const float* __restrict__ projb,  // [128]
                      const float* __restrict__ rtab,   // [169,4]
                      const int*   __restrict__ ridx,   // [2401]
                      const _Float16* __restrict__ wh,  // f16 weights (ws), may be null
                      float* __restrict__ out)
{
  extern __shared__ unsigned char smem[];

  const int blk  = blockIdx.x;
  const int bimg = blk >> 6;      // image
  const int win  = blk & 63;      // window within image
  const int wy   = win >> 3, wx = win & 7;
  const int tid  = threadIdx.x;
  const int lane = tid & 31;
  const int wave = tid >> 5;
  const int sel  = lane >> 4;     // which 16-lane half of the wave
  const int l16  = lane & 15;

  _Float16* sx   = (_Float16*)(smem + OFF_X);
  _Float16* sqkv = (_Float16*)(smem + OFF_QKV);
  _Float16* svt  = (_Float16*)(smem + OFF_VT);
  float*    ss   = (float*)(smem + OFF_S);
  int*      rid  = (int*)(smem + OFF_RID);
  float*    rsum = (float*)(smem + OFF_RS);

  // ---------------- Phase 0: region ids + shifted-window gather (f32 -> f16 LDS) ----
  if (tid < 64) {
    int v = -1;
    if (tid < NTOK) {
      int ty = tid / WSZ, tx = tid - ty * WSZ;
      int hs = wy * WSZ + ty, ws = wx * WSZ + tx;
      int rh = (hs < HH - WSZ) ? 0 : ((hs < HH - SSH) ? 1 : 2);
      int rw = (ws < WW - WSZ) ? 0 : ((ws < WW - SSH) ? 1 : 2);
      v = rh * 3 + rw;
    }
    rid[tid] = v;
  }
  for (int i = tid; i < 64 * 32; i += 256) {   // 64 rows x 32 float4 chunks
    int row = i >> 5;
    int c0  = (i & 31) << 2;
    v4f val = {0.f, 0.f, 0.f, 0.f};
    if (row < NTOK) {
      int ty = row / WSZ, tx = row - ty * WSZ;
      int gh = (wy * WSZ + ty + SSH) % HH;     // roll(x, -SS): xs[p] = x[(p+SS)%H]
      int gw = (wx * WSZ + tx + SSH) % WW;
      val = *(const v4f*)(x + (((size_t)bimg * HH + gh) * WW + gw) * CCH + c0);
    }
    _Float16* d = sx + row * CCH + c0;
    d[0] = (_Float16)val[0]; d[1] = (_Float16)val[1];
    d[2] = (_Float16)val[2]; d[3] = (_Float16)val[3];
  }
  __syncthreads();

  // ---------------- Phase 1: qkv GEMM  [64x128] @ [128x384]  (B from L2) -----------
  const float qscale = 0.17677669529663687f;   // 1/sqrt(32)
  for (int cti = 0; cti < 3; ++cti) {
    int ct  = wave * 3 + cti;                  // 24 column tiles / 8 waves
    int col = ct * 16 + l16;                   // 0..383
    v8f acc[4] = {};
#pragma unroll
    for (int kt = 0; kt < 4; ++kt) {           // K = 128 in 32-chunks
      v16h bf;
      if constexpr (F16W) bf = ldB16(wh + col * CCH + kt * 32 + sel * 16);
      else                bf = ldBf32(qkvw + col * CCH + kt * 32 + sel * 16);
#pragma unroll
      for (int r = 0; r < 4; ++r) {
        v16h af = ldsA(sx + (r * 16 + l16) * CCH, kt * 32 + sel * 8);
        acc[r] = wmma_f16(af, bf, acc[r]);
      }
    }
    float bias = qkvb[col];
    bool isq = (ct < 8);                       // q columns 0..127 get the scale
#pragma unroll
    for (int r = 0; r < 4; ++r) {
#pragma unroll
      for (int g = 0; g < 8; ++g) {
        int row = r * 16 + sel * 8 + g;
        float v = acc[r][g] + bias;
        if (isq) v *= qscale;
        _Float16 hv = (_Float16)v;
        sqkv[row * 384 + col] = hv;
        if (ct >= 16) svt[(col - 256) * 64 + row] = hv;  // V^T[h*32+d][m]
      }
    }
  }
  __syncthreads();

  // ---------------- Phase 2: S = q @ k^T per head (K = hd = 32, one WMMA/tile) -----
#pragma unroll
  for (int i = 0; i < 8; ++i) {
    int t  = wave * 8 + i;                     // 64 tiles: 4 heads x 4x4
    int h  = t >> 4;
    int rt = (t >> 2) & 3;
    int c2 = t & 3;
    v16h af = ldsA(sqkv + (rt * 16 + l16) * 384 + h * 32, sel * 8);
    int m = c2 * 16 + l16;
    v16h bf = ldB16(sqkv + m * 384 + 128 + h * 32 + sel * 16); // k rows are B^T
    v8f acc = {};
    acc = wmma_f16(af, bf, acc);
#pragma unroll
    for (int g = 0; g < 8; ++g) {
      int n = rt * 16 + sel * 8 + g;
      float s = -10000.0f;                     // pad: finite large-negative
      if (n < NTOK && m < NTOK) {
        float bias = rtab[ridx[n * NTOK + m] * NHEAD + h];
        float msk  = (rid[n] == rid[m]) ? 0.0f : -100.0f;
        s = acc[g] + bias + msk;
      }
      ss[(h * 64 + n) * 64 + m] = s;
    }
  }
  __syncthreads();

  // ---------------- Phase 3: softmax, one row per thread (256 rows total) ----------
  {
    int h = tid >> 6, n = tid & 63;
    volatile float* rp = ss + (h * 64 + n) * 64;
    float mx = -1e30f;
    for (int m2 = 0; m2 < 64; ++m2) { float s = rp[m2]; mx = fmaxf(mx, s); }
    // write unnormalized exp() as f16 in-place (2*m write trails 4*m read; volatile
    // prevents type-based reordering). Normalization is folded into the O store.
    volatile unsigned short* pp = (volatile unsigned short*)(ss + (h * 64 + n) * 64);
    float sum = 0.0f;
    for (int m2 = 0; m2 < 64; ++m2) {
      float e = __expf(rp[m2] - mx);
      sum += e;
      union { _Float16 hf; unsigned short us; } cv; cv.hf = (_Float16)e;
      pp[m2] = cv.us;
    }
    rsum[tid] = sum;
  }
  __syncthreads();

  // ---------------- Phase 4: O = P @ V per head (K = 64 -> 2 WMMA/tile) ------------
  _Float16* so = sx;                            // reuse x-window LDS for O [64x128]
  const _Float16* sp = (const _Float16*)ss;     // P rows: stride 128 halfs, 64 valid
#pragma unroll
  for (int i = 0; i < 4; ++i) {
    int t  = wave * 4 + i;                      // 32 tiles: 4 heads x 4 x 2
    int h  = t >> 3;
    int rt = (t >> 1) & 3;
    int dt = t & 1;
    v8f acc = {};
#pragma unroll
    for (int kt = 0; kt < 2; ++kt) {
      v16h af = ldsA(sp + (h * 64 + rt * 16 + l16) * 128, kt * 32 + sel * 8);
      v16h bf = ldB16(svt + (h * 32 + dt * 16 + l16) * 64 + kt * 32 + sel * 16);
      acc = wmma_f16(af, bf, acc);
    }
#pragma unroll
    for (int g = 0; g < 8; ++g) {
      int n = rt * 16 + sel * 8 + g;
      float inv = 1.0f / rsum[h * 64 + n];      // softmax normalization here
      so[n * CCH + h * 32 + dt * 16 + l16] = (_Float16)(acc[g] * inv);
    }
  }
  __syncthreads();

  // ---------------- Phase 5: proj GEMM [64x128]@[128x128] + reverse-shift store ----
  {
    int col = wave * 16 + l16;                  // each wave owns one column tile
    v8f acc[4] = {};
#pragma unroll
    for (int kt = 0; kt < 4; ++kt) {
      v16h bf;
      if constexpr (F16W) bf = ldB16(wh + QKVW_ELTS + col * CCH + kt * 32 + sel * 16);
      else                bf = ldBf32(projw + col * CCH + kt * 32 + sel * 16);
#pragma unroll
      for (int r = 0; r < 4; ++r) {
        v16h af = ldsA(so + (r * 16 + l16) * CCH, kt * 32 + sel * 8);
        acc[r] = wmma_f16(af, bf, acc[r]);
      }
    }
    float bias = projb[col];
#pragma unroll
    for (int r = 0; r < 4; ++r) {
#pragma unroll
      for (int g = 0; g < 8; ++g) {
        int n = r * 16 + sel * 8 + g;
        if (n < NTOK) {
          int ty = n / WSZ, tx = n - ty * WSZ;
          int gh = (wy * WSZ + ty + SSH) % HH;  // window-reverse + roll(+SS)
          int gw = (wx * WSZ + tx + SSH) % WW;
          out[(((size_t)bimg * HH + gh) * WW + gw) * CCH + col] = acc[r][g] + bias;
        }
      }
    }
  }
}

extern "C" void kernel_launch(void* const* d_in, const int* in_sizes, int n_in,
                              void* d_out, int out_size, void* d_ws, size_t ws_size,
                              hipStream_t stream) {
  (void)in_sizes; (void)n_in; (void)out_size;
  const float* x    = (const float*)d_in[0];
  const float* qkvw = (const float*)d_in[1];
  const float* qkvb = (const float*)d_in[2];
  const float* pw   = (const float*)d_in[3];
  const float* pb   = (const float*)d_in[4];
  const float* rtab = (const float*)d_in[5];
  const int*   ridx = (const int*)d_in[6];
  float* out = (float*)d_out;

  const bool use_f16w = (d_ws != nullptr) && (ws_size >= WS_NEEDED);

  if (use_f16w) {
    _Float16* wh = (_Float16*)d_ws;
    convert_weights_kernel<<<dim3((QKVW_ELTS + 255) / 256), dim3(256), 0, stream>>>(
        qkvw, pw, wh);
    (void)hipFuncSetAttribute((const void*)swin_attn_kernel<true>,
                              hipFuncAttributeMaxDynamicSharedMemorySize, SMEM_BYTES);
    swin_attn_kernel<true><<<dim3(64 * 64), dim3(256), SMEM_BYTES, stream>>>(
        x, qkvw, qkvb, pw, pb, rtab, ridx, wh, out);
  } else {
    (void)hipFuncSetAttribute((const void*)swin_attn_kernel<false>,
                              hipFuncAttributeMaxDynamicSharedMemorySize, SMEM_BYTES);
    swin_attn_kernel<false><<<dim3(64 * 64), dim3(256), SMEM_BYTES, stream>>>(
        x, qkvw, qkvb, pw, pb, rtab, ridx, nullptr, out);
  }
}
